// BiAttention_38585986188026
// MI455X (gfx1250) — matrix operand order repaired and checked by reading, not compiled, not run on hardware
//
#include <hip/hip_runtime.h>
#include <math.h>
#include <stdint.h>

typedef __attribute__((ext_vector_type(8)))  float  v8f;
typedef __attribute__((ext_vector_type(16))) __bf16 v16bf;

#define BB 16
#define SS 8192
#define HH 256
#define K2 (2 * HH)        // 512 : reduction dim of the main GEMM
#define WSTRIDE (3 * HH)   // 768 : row stride of attn_w
#define KD (K2 / 2)        // 256 : dword-packed K (2 bf16 per dword)
#define ROW 260            // padded LDS row stride in dwords (bank-friendly)
#define SROWS 32           // s-rows per block (2 subtiles of 16)
#define LDS_TILE (SROWS * ROW)

// ---- round-to-nearest-even fp32 -> bf16 split helpers ---------------------
__device__ __forceinline__ void split_bf16(float x, uint32_t& hbits, float& lo) {
    uint32_t u = __float_as_uint(x);
    uint32_t r = (u + 0x7fffu + ((u >> 16) & 1u)) & 0xffff0000u; // rne bf16, hi-aligned
    hbits = r;
    lo = x - __uint_as_float(r);
}
__device__ __forceinline__ uint32_t pack_bf16_rne(float x, float y) {
    uint32_t ux = __float_as_uint(x), uy = __float_as_uint(y);
    uint32_t rx = (ux + 0x7fffu + ((ux >> 16) & 1u)) >> 16;
    uint32_t ry = (uy + 0x7fffu + ((uy >> 16) & 1u)) & 0xffff0000u;
    return (rx & 0xffffu) | ry;
}

// branch-free tanh: 1 - 2/(e^{2x}+1); v_exp + v_rcp (TRANS), exact saturation
__device__ __forceinline__ float tanh_fast(float x) {
    float e = __expf(2.0f * x);
    return 1.0f - 2.0f * __builtin_amdgcn_rcpf(e + 1.0f);
}

union FragBF { v16bf v; uint4 q[2]; };

// hi*hi + hi*lo + lo*hi  (bf16x3 split product) into accumulator c
__device__ __forceinline__ void mma3(v8f& c, const uint32_t* aHi, const uint32_t* aLo,
                                     int ab, const FragBF& bhi, const FragBF& blo) {
    FragBF ahi, alo;
    ahi.q[0] = *(const uint4*)&aHi[ab];
    ahi.q[1] = *(const uint4*)&aHi[ab + 8];
    alo.q[0] = *(const uint4*)&aLo[ab];
    alo.q[1] = *(const uint4*)&aLo[ab + 8];
    c = __builtin_amdgcn_wmma_f32_16x16x32_bf16(false, ahi.v, false, bhi.v, (short)0, c, false, false);
    c = __builtin_amdgcn_wmma_f32_16x16x32_bf16(false, ahi.v, false, blo.v, (short)0, c, false, false);
    c = __builtin_amdgcn_wmma_f32_16x16x32_bf16(false, alo.v, false, bhi.v, (short)0, c, false, false);
}

// ---------------------------------------------------------------------------
// Phase 0a: hb[b,h] = attn_b[h] + sum_k hidden[b,k] * attn_w[h,k]  (fp32)
// ---------------------------------------------------------------------------
__global__ __launch_bounds__(256) void hb_kernel(
    const float* __restrict__ hidden, const float* __restrict__ attn_w,
    const float* __restrict__ attn_b, float* __restrict__ hb)
{
    int idx = blockIdx.x * 256 + threadIdx.x;   // b*H + h
    int b = idx >> 8, h = idx & (HH - 1);
    const float* hp = hidden + b * HH;
    const float* wp = attn_w + (size_t)h * WSTRIDE;
    float s = attn_b[h];
#pragma unroll 8
    for (int k = 0; k < HH; ++k) s += hp[k] * wp[k];
    hb[idx] = s;
}

// ---------------------------------------------------------------------------
// Phase 0b: split W2 = attn_w[:, H:3H] into bf16 hi/lo, dword-packed K-pairs.
// ---------------------------------------------------------------------------
__global__ __launch_bounds__(256) void w2split_kernel(
    const float* __restrict__ attn_w, uint32_t* __restrict__ w2hi,
    uint32_t* __restrict__ w2lo)
{
    int idx = blockIdx.x * 256 + threadIdx.x;   // h*KD + d
    int h = idx >> 8, d = idx & (KD - 1);
    const float* wp = attn_w + (size_t)h * WSTRIDE + HH + 2 * d;
    float x = wp[0], y = wp[1];
    uint32_t hx, hy; float lx, ly;
    split_bf16(x, hx, lx);
    split_bf16(y, hy, ly);
    w2hi[idx] = (hx >> 16) | hy;        // low K in [15:0], high K in [31:16]
    w2lo[idx] = pack_bf16_rne(lx, ly);
}

// ---------------------------------------------------------------------------
// Phase 1: per (b, 32-row s-tile):
//   E[s,h] = enc[b,s,:] . W2[h,:]  via bf16x3 split WMMA (fp32-grade accuracy)
//   ae[b,s] = sum_h v[h] * tanh(E[s,h] + hb[b,h])
// Block = 8 waves; wave w owns h-tiles 2w, 2w+1; each B fragment (register-
// resident) is reused across 2 M-subtiles -> W2/L2 traffic halved vs 16-row.
// ---------------------------------------------------------------------------
__global__ __launch_bounds__(256) void energy_kernel(
    const float* __restrict__ enc,        // [B,S,2H] fp32
    const uint32_t* __restrict__ w2hi,    // [H,KD] packed bf16
    const uint32_t* __restrict__ w2lo,    // [H,KD] packed bf16
    const float* __restrict__ hb,         // [B,H]
    const float* __restrict__ v,          // [H]
    float* __restrict__ ae)               // [B,S]
{
    extern __shared__ uint32_t smem[];
    uint32_t* sAhi = smem;                       // 32*260 dwords
    uint32_t* sAlo = smem + LDS_TILE;            // 32*260 dwords
    float*    sred = (float*)(smem + 2 * LDS_TILE); // 8*32 floats

    const int b    = blockIdx.y;
    const int s0   = blockIdx.x * SROWS;
    const int tid  = threadIdx.x;
    const int lane = tid & 31;            // wave32
    const int wave = tid >> 5;

    // ---- stage enc[s0:s0+32, 0:512]: fp32 -> bf16 hi/lo into LDS ----------
    const float4* src4 = (const float4*)(enc + ((size_t)b * SS + s0) * K2);
#pragma unroll
    for (int i = 0; i < (SROWS * K2 / 4) / 256; ++i) {   // 16 iters
        int idx4 = tid + i * 256;            // float4 index in tile
        int s = idx4 >> 7, j = idx4 & 127;   // row, float4-within-row
        float4 f = src4[idx4];
        uint32_t h0, h1, h2, h3; float l0, l1, l2, l3;
        split_bf16(f.x, h0, l0); split_bf16(f.y, h1, l1);
        split_bf16(f.z, h2, l2); split_bf16(f.w, h3, l3);
        int dst = s * ROW + 2 * j;
        *(uint2*)&sAhi[dst] = make_uint2((h0 >> 16) | h1, (h2 >> 16) | h3);
        *(uint2*)&sAlo[dst] = make_uint2(pack_bf16_rne(l0, l1), pack_bf16_rne(l2, l3));
    }
    __syncthreads();

    const int mn  = lane & 15;            // A: M row / B: N col
    const int hlf = lane >> 4;            // lane half selects K sub-range

    float accv0[8], accv1[8];
#pragma unroll
    for (int r = 0; r < 8; ++r) { accv0[r] = 0.f; accv1[r] = 0.f; }

#pragma unroll
    for (int t = 0; t < 2; ++t) {
        const int h = (wave * 2 + t) * 16 + mn;           // this lane's h col
        const uint32_t* bhiRow = w2hi + h * KD;
        const uint32_t* bloRow = w2lo + h * KD;
        const uint32_t* aHi0 = sAhi + mn * ROW;           // subtile 0 row
        const uint32_t* aLo0 = sAlo + mn * ROW;
        const uint32_t* aHi1 = aHi0 + 16 * ROW;           // subtile 1 row
        const uint32_t* aLo1 = aLo0 + 16 * ROW;

        v8f c0 = {0.f,0.f,0.f,0.f,0.f,0.f,0.f,0.f};
        v8f c1 = {0.f,0.f,0.f,0.f,0.f,0.f,0.f,0.f};
#pragma unroll 4
        for (int d0 = 0; d0 < KD; d0 += 16) {             // K chunk of 32
            // B frags: lanes 0-15: dwords d0+{0..3,8..11}; 16-31: +{4..7,12..15}
            FragBF bhi, blo;
            int bb = d0 + (hlf << 2);
            bhi.q[0] = *(const uint4*)&bhiRow[bb];
            bhi.q[1] = *(const uint4*)&bhiRow[bb + 8];
            blo.q[0] = *(const uint4*)&bloRow[bb];
            blo.q[1] = *(const uint4*)&bloRow[bb + 8];
            // A frags: lanes 0-15: dwords d0+{0..3,8..11}; 16-31: +{4..7,12..15}
            int ab = d0 + (hlf << 2);
            mma3(c0, aHi0, aLo0, ab, bhi, blo);   // B reused from registers
            mma3(c1, aHi1, aLo1, ab, bhi, blo);
        }

        // fuse bias + tanh + v-dot (lane holds fixed column h; VGPR r -> M=r|r+8)
        const float hbv = hb[b * HH + h];
        const float vv  = v[h];
#pragma unroll
        for (int r = 0; r < 8; ++r) {
            accv0[r] += vv * tanh_fast(c0[r] + hbv);
            accv1[r] += vv * tanh_fast(c1[r] + hbv);
        }
    }

    // reduce over the 16 N-lanes inside each half-wave
#pragma unroll
    for (int r = 0; r < 8; ++r) {
        float x = accv0[r], y = accv1[r];
        x += __shfl_xor(x, 1, 32); y += __shfl_xor(y, 1, 32);
        x += __shfl_xor(x, 2, 32); y += __shfl_xor(y, 2, 32);
        x += __shfl_xor(x, 4, 32); y += __shfl_xor(y, 4, 32);
        x += __shfl_xor(x, 8, 32); y += __shfl_xor(y, 8, 32);
        accv0[r] = x; accv1[r] = y;
    }
    if ((lane & 15) == 0) {
        const int mbase = hlf * 8;      // lane 0 -> m 0..7, lane 16 -> m 8..15
#pragma unroll
        for (int r = 0; r < 8; ++r) {
            sred[wave * 32 + mbase + r]      = accv0[r];
            sred[wave * 32 + 16 + mbase + r] = accv1[r];
        }
    }
    __syncthreads();

    // deterministic cross-wave sum (each wave covered disjoint h-tiles)
    if (tid < SROWS) {
        float s = 0.f;
#pragma unroll
        for (int w = 0; w < 8; ++w) s += sred[w * 32 + tid];
        ae[(size_t)b * SS + s0 + tid] = s;
    }
}

// ---------------------------------------------------------------------------
// Phase 2: softmax over S per batch.
// ---------------------------------------------------------------------------
__global__ __launch_bounds__(1024) void softmax_kernel(
    const float* __restrict__ ae, float* __restrict__ out)
{
    __shared__ float red[1024];
    const int b = blockIdx.x, tid = threadIdx.x;
    const float* row = ae + (size_t)b * SS;
    float* orow = out + (size_t)b * SS;

    float m = -INFINITY;
    for (int s = tid; s < SS; s += 1024) m = fmaxf(m, row[s]);
    red[tid] = m; __syncthreads();
    for (int off = 512; off > 0; off >>= 1) {
        if (tid < off) red[tid] = fmaxf(red[tid], red[tid + off]);
        __syncthreads();
    }
    const float mx = red[0]; __syncthreads();

    float sum = 0.f;
    for (int s = tid; s < SS; s += 1024) {
        float e = expf(row[s] - mx);
        orow[s] = e;
        sum += e;
    }
    red[tid] = sum; __syncthreads();
    for (int off = 512; off > 0; off >>= 1) {
        if (tid < off) red[tid] += red[tid + off];
        __syncthreads();
    }
    const float inv = 1.f / red[0];
    for (int s = tid; s < SS; s += 1024) orow[s] *= inv;
}

// ---------------------------------------------------------------------------
extern "C" void kernel_launch(void* const* d_in, const int* in_sizes, int n_in,
                              void* d_out, int out_size, void* d_ws, size_t ws_size,
                              hipStream_t stream)
{
    const float* hidden = (const float*)d_in[0];  // [B,1,H]
    const float* enc    = (const float*)d_in[1];  // [B,S,2H]
    const float* attn_w = (const float*)d_in[2];  // [H,3H]
    const float* attn_b = (const float*)d_in[3];  // [H]
    const float* v      = (const float*)d_in[4];  // [H]

    char* ws = (char*)d_ws;
    float*    hb   = (float*)ws;                               // B*H*4   = 16 KB
    float*    ae   = (float*)(ws + 16384);                     // B*S*4   = 512 KB
    uint32_t* w2hi = (uint32_t*)(ws + 16384 + 524288);         // H*KD*4  = 256 KB
    uint32_t* w2lo = (uint32_t*)(ws + 16384 + 524288 + 262144);// H*KD*4  = 256 KB

    const size_t lds_bytes = (2 * LDS_TILE) * sizeof(uint32_t) + 8 * 32 * sizeof(float);

    hb_kernel<<<(BB * HH) / 256, 256, 0, stream>>>(hidden, attn_w, attn_b, hb);
    w2split_kernel<<<(HH * KD) / 256, 256, 0, stream>>>(attn_w, w2hi, w2lo);
    energy_kernel<<<dim3(SS / SROWS, BB), 256, lds_bytes, stream>>>(enc, w2hi, w2lo, hb, v, ae);
    softmax_kernel<<<BB, 1024, 0, stream>>>(ae, (float*)d_out);
}
